// MemoryUnit_46712064311884
// MI455X (gfx1250) — matrix operand order repaired and checked by reading, not compiled, not run on hardware
//
#include <hip/hip_runtime.h>

typedef __attribute__((ext_vector_type(16))) __bf16          v16bf;
typedef __attribute__((ext_vector_type(8)))  float           v8f;
typedef __attribute__((ext_vector_type(16))) unsigned short  u16x16;
typedef __attribute__((ext_vector_type(8)))  unsigned short  u16x8;

union Frag16 {               // 32 bytes: one WMMA bf16 operand fragment
    u16x8  h2[2];
    u16x16 full;
    v16bf  bf;
};

#define SHRINK_THRES 0.0025f
#define EPS 1e-12f

__device__ __forceinline__ unsigned short bf16_bits(float f) {
    return __builtin_bit_cast(unsigned short, (__bf16)f);   // HW cvt (RNE)
}

// ---------------------------------------------------------------------------
// Conversion kernels (run once; W/x are re-read 100s of times by the GEMMs)
// ---------------------------------------------------------------------------
__global__ void cvt_f32_bf16(const float* __restrict__ in,
                             unsigned short* __restrict__ out, int n) {
    int i = (int)(blockIdx.x * blockDim.x + threadIdx.x) * 2;
    if (i + 1 < n) {
        float2 f = *(const float2*)(in + i);
        unsigned int p = (unsigned int)bf16_bits(f.x) |
                         ((unsigned int)bf16_bits(f.y) << 16);
        *(unsigned int*)(out + i) = p;
    } else if (i < n) {
        out[i] = bf16_bits(in[i]);
    }
}

// wt[c * Mpad + m] = bf16(W[m * C + c]), zero-padded for m >= M
__global__ void cvt_transpose_W(const float* __restrict__ W,
                                unsigned short* __restrict__ wt,
                                int M, int C, int Mpad) {
    int idx = (int)(blockIdx.x * blockDim.x + threadIdx.x);
    if (idx >= C * Mpad) return;
    int c = idx / Mpad, m = idx - c * Mpad;
    float v = (m < M) ? W[(size_t)m * C + c] : 0.0f;
    wt[idx] = bf16_bits(v);
}

// ---------------------------------------------------------------------------
// GEMM1 (fast): pre[N,M] = x @ W^T, bf16 operands from ws.
// One wave per 16x80 strip: A fragment reused across 5 WMMAs.
// ---------------------------------------------------------------------------
#define TJ 5
__global__ void gemm1_wmma(const unsigned short* __restrict__ xb,
                           const unsigned short* __restrict__ wb,
                           float* __restrict__ pre, int N, int M, int C) {
    int wave = (int)((blockIdx.x * blockDim.x + threadIdx.x) >> 5);
    int lane = (int)(threadIdx.x & 31);
    int strips = M / (16 * TJ);                  // 25
    int ti = wave / strips;
    int ts = wave - ti * strips;
    if (ti >= (N >> 4)) return;
    int i0 = ti << 4, j0 = ts * 16 * TJ;
    int h = lane >> 4, l16 = lane & 15;

    const unsigned short* arow = xb + (size_t)(i0 + l16) * C;
    const unsigned short* brow[TJ];
#pragma unroll
    for (int t = 0; t < TJ; ++t)
        brow[t] = wb + (size_t)(j0 + t * 16 + l16) * C;

    v8f acc[TJ];
#pragma unroll
    for (int t = 0; t < TJ; ++t) acc[t] = (v8f){};

    for (int kb = 0; kb < C; kb += 32) {
        Frag16 a;
        a.h2[0] = *(const u16x8*)(arow + kb + h * 16);       // K = h*16+0..7
        a.h2[1] = *(const u16x8*)(arow + kb + h * 16 + 8);   // K = h*16+8..15
#pragma unroll
        for (int t = 0; t < TJ; ++t) {
            Frag16 b;
            b.full = *(const u16x16*)(brow[t] + kb + h * 16); // K = h*16+0..15
            acc[t] = __builtin_amdgcn_wmma_f32_16x16x32_bf16(
                false, a.bf, false, b.bf, (short)0, acc[t], false, false);
        }
    }
#pragma unroll
    for (int t = 0; t < TJ; ++t) {
        float* prow = pre + (size_t)(i0 + h * 8) * M + j0 + t * 16 + l16;
#pragma unroll
        for (int r = 0; r < 8; ++r) prow[(size_t)r * M] = acc[t][r];
    }
}

// ---------------------------------------------------------------------------
// GEMM2 (fast): out[N,C] = att @ W, B from transposed bf16 W^T (zero-padded),
// A converted f32->bf16 in-register. One wave per 16x64 strip.
// ---------------------------------------------------------------------------
#define TN 4
__global__ void gemm2_wmma(const float* __restrict__ att,
                           const unsigned short* __restrict__ wt,
                           float* __restrict__ out,
                           int N, int M, int C, int Mpad) {
    int wave = (int)((blockIdx.x * blockDim.x + threadIdx.x) >> 5);
    int lane = (int)(threadIdx.x & 31);
    int strips = C / (16 * TN);                  // 4
    int ti = wave / strips;
    int ts = wave - ti * strips;
    if (ti >= (N >> 4)) return;
    int i0 = ti << 4, n0 = ts * 16 * TN;
    int h = lane >> 4, l16 = lane & 15;

    const float* arow = att + (size_t)(i0 + l16) * M;
    const unsigned short* brow[TN];
#pragma unroll
    for (int t = 0; t < TN; ++t)
        brow[t] = wt + (size_t)(n0 + t * 16 + l16) * Mpad;

    v8f acc[TN];
#pragma unroll
    for (int t = 0; t < TN; ++t) acc[t] = (v8f){};

    for (int kb = 0; kb < Mpad; kb += 32) {
        float f[16];
        if (kb + 32 <= M) {                      // full chunk: vector loads
            float4 c0 = *(const float4*)(arow + kb + h * 8);
            float4 c1 = *(const float4*)(arow + kb + h * 8 + 4);
            float4 c2 = *(const float4*)(arow + kb + 16 + h * 8);
            float4 c3 = *(const float4*)(arow + kb + 16 + h * 8 + 4);
            f[0]=c0.x; f[1]=c0.y; f[2]=c0.z; f[3]=c0.w;
            f[4]=c1.x; f[5]=c1.y; f[6]=c1.z; f[7]=c1.w;
            f[8]=c2.x; f[9]=c2.y; f[10]=c2.z; f[11]=c2.w;
            f[12]=c3.x; f[13]=c3.y; f[14]=c3.z; f[15]=c3.w;
        } else {                                 // K tail: guarded
#pragma unroll
            for (int e = 0; e < 8; ++e) {
                int k0 = kb + h * 8 + e;
                int k1 = kb + 16 + h * 8 + e;
                f[e]     = (k0 < M) ? arow[k0] : 0.0f;
                f[8 + e] = (k1 < M) ? arow[k1] : 0.0f;
            }
        }
        Frag16 a;
#pragma unroll
        for (int e = 0; e < 16; ++e) a.full[e] = bf16_bits(f[e]);
#pragma unroll
        for (int t = 0; t < TN; ++t) {
            Frag16 b;
            b.h2[0] = *(const u16x8*)(brow[t] + kb + h * 8);
            b.h2[1] = *(const u16x8*)(brow[t] + kb + 16 + h * 8);
            acc[t] = __builtin_amdgcn_wmma_f32_16x16x32_bf16(
                false, a.bf, false, b.bf, (short)0, acc[t], false, false);
        }
    }
#pragma unroll
    for (int t = 0; t < TN; ++t) {
        float* orow = out + (size_t)(i0 + h * 8) * C + n0 + t * 16 + l16;
#pragma unroll
        for (int r = 0; r < 8; ++r) orow[(size_t)r * C] = acc[t][r];
    }
}

// ---------------------------------------------------------------------------
// Row-wise softmax -> hard-shrink-relu -> L1 normalize. One block per row.
// ---------------------------------------------------------------------------
__device__ __forceinline__ float block_reduce(float v, float* sred, int is_sum) {
#pragma unroll
    for (int off = 16; off; off >>= 1) {
        float o = __shfl_xor(v, off, 32);
        v = is_sum ? (v + o) : fmaxf(v, o);
    }
    int warp = (int)(threadIdx.x >> 5), lane = (int)(threadIdx.x & 31);
    if (lane == 0) sred[warp] = v;
    __syncthreads();
    int nw = (int)(blockDim.x >> 5);
    if (warp == 0) {
        v = (lane < nw) ? sred[lane] : (is_sum ? 0.0f : -3.4e38f);
#pragma unroll
        for (int off = 16; off; off >>= 1) {
            float o = __shfl_xor(v, off, 32);
            v = is_sum ? (v + o) : fmaxf(v, o);
        }
        if (lane == 0) sred[0] = v;
    }
    __syncthreads();
    v = sred[0];
    __syncthreads();
    return v;
}

__global__ void softmax_shrink_l1(const float* __restrict__ pre,
                                  float* __restrict__ att, int M) {
    __shared__ float s_row[2000];
    __shared__ float s_red[32];
    const float* row_in = pre + (size_t)blockIdx.x * M;
    float* row_out = att + (size_t)blockIdx.x * M;
    int tid = (int)threadIdx.x, bs = (int)blockDim.x;

    float pmax = -3.4e38f;
    for (int i = tid; i < M; i += bs) {
        float v = row_in[i];
        s_row[i] = v;
        pmax = fmaxf(pmax, v);
    }
    __syncthreads();
    float rmax = block_reduce(pmax, s_red, 0);

    float psum = 0.0f;
    for (int i = tid; i < M; i += bs) {
        float e = __expf(s_row[i] - rmax);
        s_row[i] = e;
        psum += e;
    }
    __syncthreads();
    float rsum = block_reduce(psum, s_red, 1);
    float inv = 1.0f / rsum;

    float pl1 = 0.0f;
    for (int i = tid; i < M; i += bs) {
        float a = s_row[i] * inv;
        float s = a - SHRINK_THRES;
        float v = fmaxf(s, 0.0f) * a / (fabsf(s) + EPS);
        s_row[i] = v;
        pl1 += fabsf(v);
    }
    __syncthreads();
    float l1 = block_reduce(pl1, s_red, 1);
    float dinv = 1.0f / fmaxf(l1, EPS);

    for (int i = tid; i < M; i += bs) row_out[i] = s_row[i] * dinv;
}

// ---------------------------------------------------------------------------
// Fallback GEMMs (no workspace): convert f32->bf16 on the fly.
// ---------------------------------------------------------------------------
__global__ void gemm1_fly(const float* __restrict__ x, const float* __restrict__ W,
                          float* __restrict__ pre, int N, int M, int C) {
    int wave = (int)((blockIdx.x * blockDim.x + threadIdx.x) >> 5);
    int lane = (int)(threadIdx.x & 31);
    int mtiles = M >> 4;
    int ti = wave / mtiles, tj = wave - ti * mtiles;
    if (ti >= (N >> 4)) return;
    int i0 = ti << 4, j0 = tj << 4;
    int h = lane >> 4, l16 = lane & 15;
    const float* xrow = x + (size_t)(i0 + l16) * C;
    const float* wrow = W + (size_t)(j0 + l16) * C;
    v8f acc = {};
    for (int kb = 0; kb < C; kb += 32) {
        Frag16 a, b;
#pragma unroll
        for (int v = 0; v < 8; ++v) {
            int ka = kb + ((v < 4) ? 0 : 16) + h * 8 + ((v & 3) << 1);
            float2 fa = *(const float2*)(xrow + ka);
            a.full[2 * v] = bf16_bits(fa.x); a.full[2 * v + 1] = bf16_bits(fa.y);
            int kv = kb + ((v < 4) ? 0 : 16) + h * 8 + ((v & 3) << 1);
            float2 fb = *(const float2*)(wrow + kv);
            b.full[2 * v] = bf16_bits(fb.x); b.full[2 * v + 1] = bf16_bits(fb.y);
        }
        acc = __builtin_amdgcn_wmma_f32_16x16x32_bf16(
            false, a.bf, false, b.bf, (short)0, acc, false, false);
    }
    float* prow = pre + (size_t)(i0 + h * 8) * M + j0 + l16;
#pragma unroll
    for (int r = 0; r < 8; ++r) prow[(size_t)r * M] = acc[r];
}

__global__ void gemm2_fly(const float* __restrict__ att, const float* __restrict__ W,
                          float* __restrict__ out, int N, int M, int C) {
    int wave = (int)((blockIdx.x * blockDim.x + threadIdx.x) >> 5);
    int lane = (int)(threadIdx.x & 31);
    int ctiles = C >> 4;
    int ti = wave / ctiles, tn = wave - ti * ctiles;
    if (ti >= (N >> 4)) return;
    int i0 = ti << 4, n0 = tn << 4;
    int h = lane >> 4, l16 = lane & 15;
    const float* arow = att + (size_t)(i0 + l16) * M;
    const float* wcol = W + n0 + l16;
    v8f acc = {};
    int KC = (M + 31) & ~31;
    for (int kb = 0; kb < KC; kb += 32) {
        Frag16 a, b;
#pragma unroll
        for (int v = 0; v < 8; ++v) {
            int ka = kb + ((v < 4) ? 0 : 16) + h * 8 + ((v & 3) << 1);
            a.full[2 * v]     = bf16_bits((ka < M)     ? arow[ka]     : 0.0f);
            a.full[2 * v + 1] = bf16_bits((ka + 1 < M) ? arow[ka + 1] : 0.0f);
            int kv = kb + ((v < 4) ? 0 : 16) + h * 8 + ((v & 3) << 1);
            b.full[2 * v]     = bf16_bits((kv < M)     ? wcol[(size_t)kv * C]       : 0.0f);
            b.full[2 * v + 1] = bf16_bits((kv + 1 < M) ? wcol[(size_t)(kv + 1) * C] : 0.0f);
        }
        acc = __builtin_amdgcn_wmma_f32_16x16x32_bf16(
            false, a.bf, false, b.bf, (short)0, acc, false, false);
    }
    float* orow = out + (size_t)(i0 + h * 8) * C + n0 + l16;
#pragma unroll
    for (int r = 0; r < 8; ++r) orow[(size_t)r * C] = acc[r];
}

// ---------------------------------------------------------------------------
extern "C" void kernel_launch(void* const* d_in, const int* in_sizes, int n_in,
                              void* d_out, int out_size, void* d_ws, size_t ws_size,
                              hipStream_t stream) {
    (void)n_in; (void)out_size;
    const float* x = (const float*)d_in[0];
    const float* W = (const float*)d_in[1];
    const int C = 256;
    const int N = in_sizes[0] / C;           // 32768
    const int M = in_sizes[1] / C;           // 2000
    const int Mpad = (M + 31) & ~31;         // 2016

    float* out = (float*)d_out;              // [N, C]
    float* att = out + (size_t)N * C;        // [N, M]
    float* pre = att + (size_t)N * M;        // [N, M] logits

    dim3 blk(256);                           // 8 waves per block

    size_t need = ((size_t)N * C + (size_t)M * C + (size_t)C * Mpad) * 2;
    bool fast = (ws_size >= need) && ((M % (16 * TJ)) == 0) && ((C % (16 * TN)) == 0);

    if (fast) {
        unsigned short* xb = (unsigned short*)d_ws;      // [N, C] bf16
        unsigned short* wb = xb + (size_t)N * C;         // [M, C] bf16
        unsigned short* wt = wb + (size_t)M * C;         // [C, Mpad] bf16 (W^T)

        cvt_f32_bf16<<<dim3((N * C / 2 + 255) / 256), blk, 0, stream>>>(x, xb, N * C);
        cvt_f32_bf16<<<dim3((M * C / 2 + 255) / 256), blk, 0, stream>>>(W, wb, M * C);
        cvt_transpose_W<<<dim3((C * Mpad + 255) / 256), blk, 0, stream>>>(W, wt, M, C, Mpad);

        int waves1 = (N / 16) * (M / (16 * TJ));
        gemm1_wmma<<<dim3((waves1 + 7) / 8), blk, 0, stream>>>(xb, wb, pre, N, M, C);

        softmax_shrink_l1<<<dim3(N), blk, 0, stream>>>(pre, att, M);

        int waves2 = (N / 16) * (C / (16 * TN));
        gemm2_wmma<<<dim3((waves2 + 7) / 8), blk, 0, stream>>>(att, wt, out, N, M, C, Mpad);
    } else {
        int waves1 = (N / 16) * (M / 16);
        gemm1_fly<<<dim3((waves1 + 7) / 8), blk, 0, stream>>>(x, W, pre, N, M, C);
        softmax_shrink_l1<<<dim3(N), blk, 0, stream>>>(pre, att, M);
        int waves2 = (N / 16) * (C / 16);
        gemm2_fly<<<dim3((waves2 + 7) / 8), blk, 0, stream>>>(att, W, out, N, M, C);
    }
}
